// OutlookAttention_79809082294676
// MI455X (gfx1250) — matrix operand order repaired and checked by reading, not compiled, not run on hardware
//
#include <hip/hip_runtime.h>

typedef __bf16 bf16;
typedef __attribute__((ext_vector_type(16))) __bf16 bf16x16;
typedef __attribute__((ext_vector_type(8)))  __bf16 bf16x8;
typedef __attribute__((ext_vector_type(8)))  float  f32x8;

// Problem constants (static per reference)
#define BATCH 16
#define HDIM  56
#define WDIM  56
#define CIN   192
#define EDIM  384
#define HHDIM 28
#define WWDIM 28
#define HEADS 12
#define HD    32
#define NA    972           // K^4 * heads
#define NA_PAD 976          // padded to 61 tiles of 16
#define M1    (BATCH*HDIM*WDIM)      // 50176
#define M2    (BATCH*HHDIM*WWDIM)    // 12544
#define SCALE 0.17677669529663687f   // 1/sqrt(32)

// ---------------------------------------------------------------------------
// fp32 -> bf16 elementwise convert
// ---------------------------------------------------------------------------
__global__ void f32_to_bf16_k(const float* __restrict__ src, bf16* __restrict__ dst, int n) {
    int i = blockIdx.x * blockDim.x + threadIdx.x;
    if (i < n) dst[i] = (bf16)src[i];
}

// ---------------------------------------------------------------------------
// 2x2 average pool (VALID, stride 2) of x, output bf16 [B,28,28,192]
// ---------------------------------------------------------------------------
__global__ void pool2x2_k(const float* __restrict__ x, bf16* __restrict__ p, int n) {
    int i = blockIdx.x * blockDim.x + threadIdx.x;
    if (i >= n) return;
    int c  = i % CIN;
    int pp = i / CIN;
    int w  = pp % WWDIM;
    int h  = (pp / WWDIM) % HHDIM;
    int b  = pp / (WWDIM * HHDIM);
    size_t base = (((size_t)b * HDIM + 2*h) * WDIM + 2*w) * CIN + c;
    float s = x[base] + x[base + CIN] + x[base + (size_t)WDIM*CIN] + x[base + (size_t)WDIM*CIN + CIN];
    p[i] = (bf16)(s * 0.25f);
}

// ---------------------------------------------------------------------------
// Pack fp32 weight [K,N] (row-major) into WMMA bf16 B-fragment order:
// idx = ((kt*ntiles + nt)*32 + lane)*16 + e
// element e of lane: Kg = kt*32 + (e>=8 ? 16:0) + (lane>=16 ? 8:0) + (e&7) ; N = nt*16 + (lane&15)
// ---------------------------------------------------------------------------
__global__ void pack_w_k(const float* __restrict__ W, bf16* __restrict__ Bp,
                         int K, int N, int ktiles, int ntiles) {
    int idx = blockIdx.x * blockDim.x + threadIdx.x;
    int total = ktiles * ntiles * 512;
    if (idx >= total) return;
    int e    = idx & 15;
    int lane = (idx >> 4) & 31;
    int t    = idx >> 9;
    int nt   = t % ntiles;
    int kt   = t / ntiles;
    int k    = kt * 32 + ((e >> 3) << 4) + ((lane >> 4) << 3) + (e & 7);
    int n    = nt * 16 + (lane & 15);
    float v  = (k < K && n < N) ? W[(size_t)k * N + n] : 0.f;
    Bp[idx] = (bf16)v;
}

// ---------------------------------------------------------------------------
// WMMA GEMM: C[M,N] = A[M,K](bf16,row-major) * Bpacked + bias
// Register-blocked: one wave computes a 64x16 C strip (4 M-tiles sharing one
// B fragment -> 4 independent WMMA accumulator chains per K-step).
// A fragment (16-bit A 16x32, ISA 7.12.2): lane<16 row=lane, K in {0..7,16..23};
//                                          lane>=16 K in {8..15,24..31}.
// ---------------------------------------------------------------------------
__global__ __launch_bounds__(256) void wmma_gemm_k(
    const bf16*  __restrict__ A,
    const bf16*  __restrict__ Bp,
    const float* __restrict__ bias,   // may be null
    float*       __restrict__ Cf,     // fp32 output (or null)
    bf16*        __restrict__ Cb,     // bf16 output (or null)
    int tilesM4, int ntiles, int ktiles, int lda, int Nreal, int ldc)
{
    int wave  = blockIdx.x * 8 + (threadIdx.x >> 5);
    int total = tilesM4 * ntiles;
    if (wave >= total) return;
    int mt4  = wave / ntiles;          // 64-row strip index
    int nt   = wave % ntiles;
    int lane = threadIdx.x & 31;
    int hi16 = lane >> 4;              // 0 | 1
    int l15  = lane & 15;

    const bf16* arow0 = A + (size_t)(mt4 * 64 + l15) * lda + (hi16 ? 8 : 0);
    size_t astep = (size_t)16 * lda;   // row stride between M-tiles

    f32x8 acc0 = {0.f,0.f,0.f,0.f,0.f,0.f,0.f,0.f};
    f32x8 acc1 = acc0, acc2 = acc0, acc3 = acc0;

    for (int kt = 0; kt < ktiles; ++kt) {
        // B fragment: shared by all 4 M-tiles this K-step
        const bf16* bp = Bp + (((size_t)kt * ntiles + nt) * 32 + lane) * 16;
        bf16x8 blo = *(const bf16x8*)(bp);
        bf16x8 bhi = *(const bf16x8*)(bp + 8);
        bf16x16 b = __builtin_shufflevector(blo, bhi,
            0,1,2,3,4,5,6,7,8,9,10,11,12,13,14,15);

        const bf16* ap = arow0 + kt * 32;
        if (kt + 1 < ktiles) {
            // prefetch next K-step's A lines (global_prefetch_b8)
            __builtin_prefetch(ap + 32, 0, 3);
            __builtin_prefetch(ap + 32 + astep, 0, 3);
            __builtin_prefetch(ap + 32 + 2 * astep, 0, 3);
            __builtin_prefetch(ap + 32 + 3 * astep, 0, 3);
        }

        bf16x8 a0l = *(const bf16x8*)(ap);
        bf16x8 a0h = *(const bf16x8*)(ap + 16);
        bf16x8 a1l = *(const bf16x8*)(ap + astep);
        bf16x8 a1h = *(const bf16x8*)(ap + astep + 16);
        bf16x8 a2l = *(const bf16x8*)(ap + 2 * astep);
        bf16x8 a2h = *(const bf16x8*)(ap + 2 * astep + 16);
        bf16x8 a3l = *(const bf16x8*)(ap + 3 * astep);
        bf16x8 a3h = *(const bf16x8*)(ap + 3 * astep + 16);

        bf16x16 a0 = __builtin_shufflevector(a0l, a0h, 0,1,2,3,4,5,6,7,8,9,10,11,12,13,14,15);
        bf16x16 a1 = __builtin_shufflevector(a1l, a1h, 0,1,2,3,4,5,6,7,8,9,10,11,12,13,14,15);
        bf16x16 a2 = __builtin_shufflevector(a2l, a2h, 0,1,2,3,4,5,6,7,8,9,10,11,12,13,14,15);
        bf16x16 a3 = __builtin_shufflevector(a3l, a3h, 0,1,2,3,4,5,6,7,8,9,10,11,12,13,14,15);

        // 4 independent accumulator chains -> WMMA co-issue, hazards hidden
        acc0 = __builtin_amdgcn_wmma_f32_16x16x32_bf16(false, a0, false, b, (short)0, acc0, false, false);
        acc1 = __builtin_amdgcn_wmma_f32_16x16x32_bf16(false, a1, false, b, (short)0, acc1, false, false);
        acc2 = __builtin_amdgcn_wmma_f32_16x16x32_bf16(false, a2, false, b, (short)0, acc2, false, false);
        acc3 = __builtin_amdgcn_wmma_f32_16x16x32_bf16(false, a3, false, b, (short)0, acc3, false, false);
    }

    int col = nt * 16 + l15;
    if (col < Nreal) {
        float bv = bias ? bias[col] : 0.f;
        f32x8 accs[4] = {acc0, acc1, acc2, acc3};
        #pragma unroll
        for (int m = 0; m < 4; ++m) {
            int rbase = mt4 * 64 + m * 16 + (hi16 ? 8 : 0);
            if (Cb) {
                #pragma unroll
                for (int r = 0; r < 8; ++r)
                    Cb[(size_t)(rbase + r) * ldc + col] = (bf16)(accs[m][r] + bv);
            } else {
                #pragma unroll
                for (int r = 0; r < 8; ++r)
                    Cf[(size_t)(rbase + r) * ldc + col] = accs[m][r] + bv;
            }
        }
    }
}

// ---------------------------------------------------------------------------
// softmax over q (9 elems), scale folded in; in-place on attn logits [M2, 976]
// one thread per (pixel, head, p)
// ---------------------------------------------------------------------------
__global__ void softmax9_k(float* __restrict__ attn, int n) {
    int i = blockIdx.x * blockDim.x + threadIdx.x;
    if (i >= n) return;
    int p    = i % 9;
    int head = (i / 9) % HEADS;
    int pp   = i / (9 * HEADS);
    float* row = attn + (size_t)pp * NA_PAD + head * 81 + p * 9;
    float z[9];
    float m = -1e30f;
    #pragma unroll
    for (int q = 0; q < 9; ++q) { z[q] = row[q] * SCALE; m = fmaxf(m, z[q]); }
    float s = 0.f;
    #pragma unroll
    for (int q = 0; q < 9; ++q) { z[q] = __expf(z[q] - m); s += z[q]; }
    float inv = 1.f / s;
    #pragma unroll
    for (int q = 0; q < 9; ++q) row[q] = z[q] * inv;
}

// ---------------------------------------------------------------------------
// Fused local aggregation + overlap-add fold (gather form), bf16 out.
// Output pixel (b,y,x), channel e gathers from windows (h,w) with
// 2h+i == y+1, 2w+j == x+1 (i,j in 0..2), each weighted by attn[head, i*3+j, q]
// over the 9 taps v[2h+qi-1, 2w+qj-1, e].
// grid = B*H*W blocks, 384 threads (one per channel).
// ---------------------------------------------------------------------------
__global__ __launch_bounds__(384) void fold_agg_k(
    const float* __restrict__ attn, const bf16* __restrict__ v, bf16* __restrict__ folded)
{
    int e   = threadIdx.x;
    int pix = blockIdx.x;
    int xo  = pix % WDIM;
    int yo  = (pix / WDIM) % HDIM;
    int b   = pix / (WDIM * HDIM);
    int head = e >> 5;

    float acc = 0.f;
    for (int i = 0; i < 3; ++i) {
        int t = yo + 1 - i;
        if (t < 0 || (t & 1)) continue;
        int h = t >> 1;
        if (h >= HHDIM) continue;
        for (int j = 0; j < 3; ++j) {
            int u = xo + 1 - j;
            if (u < 0 || (u & 1)) continue;
            int w = u >> 1;
            if (w >= WWDIM) continue;
            int pp = (b * HHDIM + h) * WWDIM + w;
            const float* arow = attn + (size_t)pp * NA_PAD + head * 81 + (i * 3 + j) * 9;
            #pragma unroll
            for (int q = 0; q < 9; ++q) {
                int vy = 2 * h + (q / 3) - 1;
                int vx = 2 * w + (q % 3) - 1;
                float vv = 0.f;
                if (vy >= 0 && vy < HDIM && vx >= 0 && vx < WDIM)
                    vv = (float)v[(((size_t)b * HDIM + vy) * WDIM + vx) * EDIM + e];
                acc += arow[q] * vv;
            }
        }
    }
    folded[(size_t)pix * EDIM + e] = (bf16)acc;
}

// ---------------------------------------------------------------------------
extern "C" void kernel_launch(void* const* d_in, const int* in_sizes, int n_in,
                              void* d_out, int out_size, void* d_ws, size_t ws_size,
                              hipStream_t stream) {
    const float* x  = (const float*)d_in[0];
    const float* Wv = (const float*)d_in[1];
    const float* Wa = (const float*)d_in[2];
    const float* ba = (const float*)d_in[3];
    const float* Wo = (const float*)d_in[4];
    const float* bo = (const float*)d_in[5];

    char* ws = (char*)d_ws;
    size_t off = 0;
    auto take = [&](size_t bytes) -> char* {
        char* p = ws + off;
        off += (bytes + 255) & ~(size_t)255;
        return p;
    };

    bf16*  x_bf    = (bf16*) take((size_t)M1 * CIN * 2);          // 19.3 MB
    bf16*  pooled  = (bf16*) take((size_t)M2 * CIN * 2);          //  4.8 MB
    bf16*  v_bf    = (bf16*) take((size_t)M1 * EDIM * 2);         // 38.5 MB
    bf16*  Wv_pk   = (bf16*) take((size_t)6  * 24 * 512 * 2);
    bf16*  Wa_pk   = (bf16*) take((size_t)6  * 61 * 512 * 2);
    bf16*  Wo_pk   = (bf16*) take((size_t)12 * 24 * 512 * 2);
    float* attn    = (float*)take((size_t)M2 * NA_PAD * 4);       // 49.0 MB
    bf16*  folded  = (bf16*) take((size_t)M1 * EDIM * 2);         // 38.5 MB
    (void)ws_size;

    // 1) x -> bf16
    {
        int n = M1 * CIN;
        f32_to_bf16_k<<<(n + 255) / 256, 256, 0, stream>>>(x, x_bf, n);
    }
    // 2) 2x2 avg pool -> bf16
    {
        int n = M2 * CIN;
        pool2x2_k<<<(n + 255) / 256, 256, 0, stream>>>(x, pooled, n);
    }
    // 3) pack weights into B-fragment order
    {
        int n = 6 * 24 * 512;
        pack_w_k<<<(n + 255) / 256, 256, 0, stream>>>(Wv, Wv_pk, CIN, EDIM, 6, 24);
        n = 6 * 61 * 512;
        pack_w_k<<<(n + 255) / 256, 256, 0, stream>>>(Wa, Wa_pk, CIN, NA, 6, 61);
        n = 12 * 24 * 512;
        pack_w_k<<<(n + 255) / 256, 256, 0, stream>>>(Wo, Wo_pk, EDIM, EDIM, 12, 24);
    }
    // 4) v = x @ Wv   (bf16 out)
    {
        int tilesM4 = M1 / 64, ntiles = 24;
        int blocks = (tilesM4 * ntiles + 7) / 8;
        wmma_gemm_k<<<blocks, 256, 0, stream>>>(x_bf, Wv_pk, nullptr,
                                                nullptr, v_bf,
                                                tilesM4, ntiles, 6, CIN, EDIM, EDIM);
    }
    // 5) attn_logits = pooled @ Wa + ba   (fp32 out, ldc padded to 976)
    {
        int tilesM4 = M2 / 64, ntiles = 61;
        int blocks = (tilesM4 * ntiles + 7) / 8;
        wmma_gemm_k<<<blocks, 256, 0, stream>>>(pooled, Wa_pk, ba,
                                                attn, nullptr,
                                                tilesM4, ntiles, 6, CIN, NA, NA_PAD);
    }
    // 6) softmax over q (in-place)
    {
        int n = M2 * HEADS * 9;
        softmax9_k<<<(n + 255) / 256, 256, 0, stream>>>(attn, n);
    }
    // 7) fused local aggregation + fold (gather) -> folded bf16
    {
        fold_agg_k<<<M1, EDIM, 0, stream>>>(attn, v_bf, folded);
    }
    // 8) out = folded @ Wo + bo  (fp32 out to d_out)
    {
        int tilesM4 = M1 / 64, ntiles = 24;
        int blocks = (tilesM4 * ntiles + 7) / 8;
        wmma_gemm_k<<<blocks, 256, 0, stream>>>(folded, Wo_pk, bo,
                                                (float*)d_out, nullptr,
                                                tilesM4, ntiles, 12, EDIM, EDIM, EDIM);
    }
}
